// SwinBlock_63685775065697
// MI455X (gfx1250) — matrix operand order repaired and checked by reading, not compiled, not run on hardware
//
#include <hip/hip_runtime.h>
#include <hip/hip_bf16.h>
#include <math.h>

// ---------------------------------------------------------------------------
// Swin block (shifted-window MHA + MLP) for gfx1250 (MI455X), wave32 + WMMA.
// B=4, C=192, H=W=224, ws=7, heads=6, hd=32, shift=3. No padding (224%7==0).
// All GEMMs use v_wmma_f32_16x16x32_f16 (f16 in, f32 acc).
// Bulk global->LDS fills use GLOBAL_LOAD_ASYNC_TO_LDS_B128 (ASYNCcnt).
// ---------------------------------------------------------------------------

typedef __attribute__((ext_vector_type(16))) _Float16 v16h;
typedef __attribute__((ext_vector_type(8)))  float    v8f;

#define C_DIM   192
#define DF_DIM  768
#define HW_IMG  224
#define HWPLANE 50176        // 224*224
#define BATCH   4
#define WSZ     7
#define NWGRID  32           // 224/7
#define LWIN    49
#define LPAD    64
#define NHEADS  6
#define HDIM    32

__device__ inline v8f zero8() {
  v8f z;
#pragma unroll
  for (int i = 0; i < 8; ++i) z[i] = 0.0f;
  return z;
}

__device__ inline v8f wmma16(v16h a, v16h b, v8f c) {
  // (neg_a, A, neg_b, B, c_mod, C, reuse_a, reuse_b)
  return __builtin_amdgcn_wmma_f32_16x16x32_f16(false, a, false, b, (short)0, c,
                                                false, false);
}

// A fragment (16x32, f16). Caller passes r = tile + m*ld + kbase + k0,
// where m = lane&15 and k0 = (lane&16)?16:0. Halves 0..15 <- K k0..k0+15.
__device__ inline v16h load_a_frag(const _Float16* r) {
  v16h a;
#pragma unroll
  for (int j = 0; j < 16; ++j) a[j] = r[j];
  return a;
}

// B fragment (32x16, f16) from [N][K] row-major storage. Caller passes
// r = base + n*ld + kbase + kb0, with n = lane&15, kb0 = (lane&16)?16:0.
// Lane holds 16 contiguous K values of its column.
__device__ inline v16h load_b_frag(const _Float16* r) {
  v16h b;
#pragma unroll
  for (int j = 0; j < 16; ++j) b[j] = r[j];
  return b;
}

// ---- CDNA5 async global->LDS (16B per lane, tracked with ASYNCcnt) --------
// LDS byte offset = low 32 bits of the generic pointer (ISA aperture rule).
__device__ inline void async_load_b128(uint32_t lds_off, const void* gptr) {
  asm volatile("global_load_async_to_lds_b128 %0, %1, off"
               :: "v"(lds_off), "v"((uint64_t)(uintptr_t)gptr)
               : "memory");
}
__device__ inline void wait_asynccnt0() {
  asm volatile("s_wait_asynccnt 0x0" ::: "memory");
}

// ---------------------------------------------------------------------------
// Weight f32 -> f16 cast ([N][K] row-major already matches WMMA B layout).
// ---------------------------------------------------------------------------
__global__ void cast_f16_kernel(const float* __restrict__ src,
                                _Float16* __restrict__ dst, int n) {
  int i = blockIdx.x * 256 + threadIdx.x;
  if (i < n) dst[i] = (_Float16)src[i];
}

// ---------------------------------------------------------------------------
// Kernel 1: NCHW -> NHWC transpose + LayerNorm1, output f16 tn[B][H][W][C].
// One block = 32 consecutive w for a fixed (b,h). 256 thr = 8 c-groups x 32 w.
// ---------------------------------------------------------------------------
__global__ __launch_bounds__(256) void ln1_kernel(
    const float* __restrict__ x, const float* __restrict__ g,
    const float* __restrict__ beta, _Float16* __restrict__ tn) {
  __shared__ float vals[C_DIM * 32];       // [c][w] 24KB
  __shared__ float psum[8 * 32], psq[8 * 32];
  __shared__ float mean_s[32], rstd_s[32];
  __shared__ _Float16 tile[32 * C_DIM];    // [w][c] 12KB

  const int tid = threadIdx.x, lane = tid & 31, cg = tid >> 5;
  int blk = blockIdx.x;                    // b*224*7 + h*7 + wt
  const int wt = blk % 7; blk /= 7;
  const int h = blk % HW_IMG;
  const int b = blk / HW_IMG;
  const int w0 = wt * 32;

  const float* xb = x + ((size_t)b * C_DIM) * HWPLANE + (size_t)h * HW_IMG + w0;
  float s = 0.f, q = 0.f;
  for (int c = cg; c < C_DIM; c += 8) {
    float v = xb[(size_t)c * HWPLANE + lane];   // coalesced in w
    vals[c * 32 + lane] = v;
    s += v; q += v * v;
  }
  psum[cg * 32 + lane] = s;
  psq[cg * 32 + lane] = q;
  __syncthreads();
  if (cg == 0) {
    float S = 0.f, Q = 0.f;
#pragma unroll
    for (int gg = 0; gg < 8; ++gg) { S += psum[gg * 32 + lane]; Q += psq[gg * 32 + lane]; }
    float mean = S / (float)C_DIM;
    float var  = Q / (float)C_DIM - mean * mean;
    mean_s[lane] = mean;
    rstd_s[lane] = rsqrtf(var + 1e-5f);
  }
  __syncthreads();
  const float mean = mean_s[lane], rstd = rstd_s[lane];
  for (int c = cg; c < C_DIM; c += 8) {
    float v = vals[c * 32 + lane];
    tile[lane * C_DIM + c] = (_Float16)((v - mean) * rstd * g[c] + beta[c]);
  }
  __syncthreads();
  // contiguous 32*192 halves -> 16B-chunk cooperative store
  _Float16* dst = tn + (((size_t)b * HW_IMG + h) * HW_IMG + w0) * C_DIM;
  const uint4* s4 = (const uint4*)tile;
  uint4* d4 = (uint4*)dst;
  for (int i = tid; i < (32 * C_DIM * 2) / 16; i += 256) d4[i] = s4[i];
}

// ---------------------------------------------------------------------------
// Kernel 2: windowed attention. One block per (b, window). 8 wave32.
// ---------------------------------------------------------------------------
__global__ __launch_bounds__(256) void swin_attn_kernel(
    const _Float16* __restrict__ tn,
    const _Float16* __restrict__ wqh, const float* __restrict__ bq,
    const _Float16* __restrict__ wkh, const float* __restrict__ bk,
    const _Float16* __restrict__ wvh, const float* __restrict__ bv,
    const _Float16* __restrict__ woh, const float* __restrict__ bo,
    float* __restrict__ attn_out) {
  extern __shared__ char smem[];
  _Float16* xw = (_Float16*)smem;          // [64][192]
  _Float16* qb = xw + LPAD * C_DIM;        // [64][192]
  _Float16* kb = qb + LPAD * C_DIM;        // [64][192]
  _Float16* vt = kb + LPAD * C_DIM;        // [192][64] (V transposed)
  _Float16* ob = vt + C_DIM * LPAD;        // [64][192] (head-concat PV out)
  _Float16* ps = ob + LPAD * C_DIM;        // 8 waves x [16][64] P scratch

  const int tid = threadIdx.x, lane = tid & 31, wid = tid >> 5;
  const int nloc = lane & 15;
  const int m0 = (lane & 16) ? 8 : 0;
  const int ka0 = (lane & 16) ? 16 : 0;    // A-fragment K base
  const int kb0 = (lane & 16) ? 16 : 0;    // B-fragment K base

  const int win = blockIdx.x & 1023;
  const int b = blockIdx.x >> 10;
  const int wh = win >> 5, ww = win & 31;

  // ---- async gather of rolled window rows (49 tokens x 384B) into LDS ----
  const uint32_t xw_base = (uint32_t)(uintptr_t)(void*)xw;
  for (int ch = tid; ch < LWIN * 24; ch += 256) {      // 24 x 16B chunks/row
    const int row = ch / 24, part = ch % 24;
    const int ti = row / WSZ, tj = row % WSZ;
    const int hi = (wh * WSZ + ti + 221) % HW_IMG;     // pre-roll coord
    const int wi = (ww * WSZ + tj + 221) % HW_IMG;
    const _Float16* src =
        tn + (((size_t)b * HW_IMG + hi) * HW_IMG + wi) * C_DIM + part * 8;
    async_load_b128(xw_base + (uint32_t)(row * C_DIM + part * 8) * 2, src);
  }
  // zero the pad rows (49..63) with plain LDS stores
  for (int idx = LWIN * C_DIM + tid; idx < LPAD * C_DIM; idx += 256)
    xw[idx] = (_Float16)0.f;
  // warm O-projection weights into cache while QKV runs (global_prefetch_b8)
  for (int p = tid; p < (C_DIM * C_DIM * 2) / 128; p += 256)
    __builtin_prefetch((const char*)woh + p * 128, 0, 0);
  wait_asynccnt0();
  __syncthreads();

  // ---- QKV projections: 3 mats x 4 m-tiles x 12 n-tiles = 144 tiles ----
  // mat loop unrolled so the store target is compile-time (no exec churn).
#pragma unroll
  for (int mat = 0; mat < 3; ++mat) {
    const _Float16* wsel = (mat == 0) ? wqh : (mat == 1) ? wkh : wvh;
    const float*    bsel = (mat == 0) ? bq  : (mat == 1) ? bk  : bv;
    for (int t = wid; t < 48; t += 8) {
      const int mt = t / 12, nt = t % 12;
      const int ng = nt * 16 + nloc;
      v8f acc = zero8();
      const _Float16* arow = xw + (mt * 16 + nloc) * C_DIM + ka0;
      const _Float16* brow = wsel + ng * C_DIM + kb0;
#pragma unroll
      for (int ks = 0; ks < 6; ++ks)
        acc = wmma16(load_a_frag(arow + ks * 32), load_b_frag(brow + ks * 32), acc);
      const float bias = bsel[ng];
#pragma unroll
      for (int i = 0; i < 8; ++i) {
        const int row = mt * 16 + m0 + i;
        const _Float16 v = (_Float16)(acc[i] + bias);
        if (mat == 0)      qb[row * C_DIM + ng] = v;
        else if (mat == 1) kb[row * C_DIM + ng] = v;
        else               vt[ng * LPAD + row] = v;    // V stored transposed
      }
    }
  }
  __syncthreads();

  // ---- per (head, m-tile): scores + ALiBi/mask + softmax + P@V ----
  _Float16* pw = ps + wid * 16 * LPAD;
  const float rscale = 0.17677669529663687f;           // 1/sqrt(32)
  for (int task = wid; task < 24; task += 8) {
    const int hh = task >> 2, mt = task & 3;
    const float slope = __powf(10.0f, -(float)hh * 0.2f);

    const _Float16* qrow = qb + (mt * 16 + nloc) * C_DIM + hh * HDIM + ka0;
    const v16h aq = load_a_frag(qrow);                 // K = hd = 32: one frag
    float sv[4][8];
#pragma unroll
    for (int nt = 0; nt < 4; ++nt) {
      const _Float16* krow = kb + (nt * 16 + nloc) * C_DIM + hh * HDIM + kb0;
      v8f c = zero8();
      c = wmma16(aq, load_b_frag(krow), c);
#pragma unroll
      for (int i = 0; i < 8; ++i) sv[nt][i] = c[i] * rscale;
    }
    // per-lane key-column metadata (col = nt*16 + nloc)
    int kvalid[4], kreg[4], kci[4], kcj[4];
#pragma unroll
    for (int nt = 0; nt < 4; ++nt) {
      const int kg = nt * 16 + nloc;
      kvalid[nt] = (kg < LWIN);
      const int ki = kg / WSZ, kj = kg % WSZ;
      kci[nt] = ki; kcj[nt] = kj;
      const int hik = (wh * WSZ + ki + 221) % HW_IMG;
      const int wik = (ww * WSZ + kj + 221) % HW_IMG;
      kreg[nt] = (wik >= 221 ? 1 : 0) + (hik >= 221 ? 2 : 0);
    }
#pragma unroll
    for (int i = 0; i < 8; ++i) {
      const int qg = mt * 16 + m0 + i;
      const int qok = (qg < LWIN);
      const int qi = qg / WSZ, qj = qg % WSZ;
      const int hiq = (wh * WSZ + qi + 221) % HW_IMG;
      const int wiq = (ww * WSZ + qj + 221) % HW_IMG;
      const int qreg = (wiq >= 221 ? 1 : 0) + (hiq >= 221 ? 2 : 0);
      float v[4];
#pragma unroll
      for (int nt = 0; nt < 4; ++nt) {
        int di = qi - kci[nt]; di = di < 0 ? -di : di;
        int dj = qj - kcj[nt]; dj = dj < 0 ? -dj : dj;
        const bool ok = qok && kvalid[nt] && (qreg == kreg[nt]);
        v[nt] = ok ? (sv[nt][i] - slope * (float)(di + dj)) : -INFINITY;
      }
      // row max / sum across the 16 lanes holding this row (xor masks < 16
      // stay within the lane half, so rows don't mix)
      float mx = fmaxf(fmaxf(v[0], v[1]), fmaxf(v[2], v[3]));
      for (int off = 1; off < 16; off <<= 1) mx = fmaxf(mx, __shfl_xor(mx, off, 32));
      float e[4], ssum = 0.f;
#pragma unroll
      for (int nt = 0; nt < 4; ++nt) {
        e[nt] = (mx > -INFINITY) ? __expf(v[nt] - mx) : 0.f;
        ssum += e[nt];
      }
      for (int off = 1; off < 16; off <<= 1) ssum += __shfl_xor(ssum, off, 32);
      const float inv = (ssum > 0.f) ? 1.f / ssum : 0.f;
#pragma unroll
      for (int nt = 0; nt < 4; ++nt)
        pw[(m0 + i) * LPAD + nt * 16 + nloc] = (_Float16)(e[nt] * inv);
    }
    // P (16x64) @ V_h (64x32): 2 n-tiles x 2 K-steps (LDS in-order per wave)
#pragma unroll
    for (int n2 = 0; n2 < 2; ++n2) {
      v8f oc = zero8();
#pragma unroll
      for (int ks = 0; ks < 2; ++ks) {
        const _Float16* pr = pw + nloc * LPAD + ks * 32 + ka0;
        const int nd = hh * HDIM + n2 * 16 + nloc;
        const _Float16* vr = vt + nd * LPAD + ks * 32 + kb0;
        oc = wmma16(load_a_frag(pr), load_b_frag(vr), oc);
      }
#pragma unroll
      for (int i = 0; i < 8; ++i)
        ob[(mt * 16 + m0 + i) * C_DIM + hh * HDIM + n2 * 16 + nloc] = (_Float16)oc[i];
    }
  }
  __syncthreads();

  // ---- output projection + un-roll scatter ----
  for (int task = wid; task < 48; task += 8) {
    const int mt = task / 12, nt = task % 12;
    const int ng = nt * 16 + nloc;
    v8f acc = zero8();
    const _Float16* arow = ob + (mt * 16 + nloc) * C_DIM + ka0;
    const _Float16* brow = woh + ng * C_DIM + kb0;
#pragma unroll
    for (int ks = 0; ks < 6; ++ks)
      acc = wmma16(load_a_frag(arow + ks * 32), load_b_frag(brow + ks * 32), acc);
    const float bias = bo[ng];
#pragma unroll
    for (int i = 0; i < 8; ++i) {
      const int t = mt * 16 + m0 + i;
      if (t < LWIN) {
        const int ti = t / WSZ, tj = t % WSZ;
        const int fh = (wh * WSZ + ti + 221) % HW_IMG;   // roll(-s)
        const int fw = (ww * WSZ + tj + 221) % HW_IMG;
        attn_out[(((size_t)b * HW_IMG + fh) * HW_IMG + fw) * C_DIM + ng] =
            acc[i] + bias;
      }
    }
  }
}

// ---------------------------------------------------------------------------
// Kernel 3: y = t + attn; LN2; MLP (192->768 GELU ->192); out = y + mlp (NCHW)
// One block = 64 consecutive tokens (never crosses batch: HW%64==0). 8 waves.
// Hidden dim processed in 2 halves of 384 so GEMM2 accumulates in registers.
// ---------------------------------------------------------------------------
__global__ __launch_bounds__(256) void mlp_kernel(
    const float* __restrict__ x, const float* __restrict__ attn,
    const float* __restrict__ g2, const float* __restrict__ be2,
    const _Float16* __restrict__ w1h, const float* __restrict__ b1,
    const _Float16* __restrict__ w2h, const float* __restrict__ b2,
    float* __restrict__ out) {
  extern __shared__ char smem[];
  float*    y  = (float*)smem;                               // [64][192] f32
  _Float16* mh = (_Float16*)(smem + LPAD * C_DIM * 4);       // [64][192] f16
  _Float16* gh = (_Float16*)(smem + LPAD * C_DIM * 6);       // [64][384] f16
  float* outT  = (float*)gh;                                 // reuse: [192][64]

  const int tid = threadIdx.x, lane = tid & 31, wid = tid >> 5;
  const int nloc = lane & 15;
  const int m0 = (lane & 16) ? 8 : 0;
  const int ka0 = (lane & 16) ? 16 : 0;
  const int kb0 = (lane & 16) ? 16 : 0;

  const int n0 = blockIdx.x * 64;
  const int b = n0 / HWPLANE;
  const int hw0 = n0 % HWPLANE;

  // ---- y = attn tile: async global->LDS bulk copy (contiguous 48KB) ----
  const uint32_t y_base = (uint32_t)(uintptr_t)(void*)y;
  const char* asrc = (const char*)(attn + (size_t)n0 * C_DIM);
  for (int c4 = tid; c4 < (LPAD * C_DIM * 4) / 16; c4 += 256)
    async_load_b128(y_base + (uint32_t)c4 * 16, asrc + (size_t)c4 * 16);
  wait_asynccnt0();
  __syncthreads();
  // ---- y += t (transpose read of x, coalesced along w) ----
  for (int idx = tid; idx < LPAD * C_DIM; idx += 256) {
    const int c = idx / 64, m = idx % 64;
    y[m * C_DIM + c] += x[((size_t)b * C_DIM + c) * HWPLANE + hw0 + m];
  }
  __syncthreads();

  // ---- LN2 -> mh (f16); one wave per 8 tokens ----
  for (int t = wid * 8; t < wid * 8 + 8; ++t) {
    float s = 0.f, q = 0.f;
    for (int c = lane; c < C_DIM; c += 32) {
      const float v = y[t * C_DIM + c];
      s += v; q += v * v;
    }
    for (int off = 1; off < 32; off <<= 1) {
      s += __shfl_xor(s, off, 32);
      q += __shfl_xor(q, off, 32);
    }
    const float mean = s / (float)C_DIM;
    const float rstd = rsqrtf(q / (float)C_DIM - mean * mean + 1e-5f);
    for (int c = lane; c < C_DIM; c += 32)
      mh[t * C_DIM + c] =
          (_Float16)((y[t * C_DIM + c] - mean) * rstd * g2[c] + be2[c]);
  }
  __syncthreads();

  // ---- MLP, hidden dim in two halves of 384 ----
  v8f acc[6];
#pragma unroll
  for (int i = 0; i < 6; ++i) acc[i] = zero8();

  for (int half = 0; half < 2; ++half) {
    // GEMM1: mh[64x192] @ w1^T half -> gelu -> gh[64x384]
    for (int task = wid; task < 96; task += 8) {
      const int mt = task / 24, nt = task % 24;
      const int f = half * 384 + nt * 16 + nloc;
      v8f c = zero8();
      const _Float16* arow = mh + (mt * 16 + nloc) * C_DIM + ka0;
      const _Float16* brow = w1h + f * C_DIM + kb0;
#pragma unroll
      for (int ks = 0; ks < 6; ++ks)
        c = wmma16(load_a_frag(arow + ks * 32), load_b_frag(brow + ks * 32), c);
      const float bb = b1[f];
#pragma unroll
      for (int i = 0; i < 8; ++i) {
        float v = c[i] + bb;
        v = 0.5f * v * (1.0f + erff(v * 0.70710678118654752f));   // exact GELU
        gh[(mt * 16 + m0 + i) * 384 + nt * 16 + nloc] = (_Float16)v;
      }
    }
    __syncthreads();
    // GEMM2 partial: gh[64x384] @ w2half^T, accumulate in registers
    for (int tk = 0; tk < 6; ++tk) {
      const int task = wid + tk * 8;
      const int mt = task / 12, nt = task % 12;
      const int ng = nt * 16 + nloc;
      const _Float16* arow = gh + (mt * 16 + nloc) * 384 + ka0;
      const _Float16* brow = w2h + ng * DF_DIM + half * 384 + kb0;
#pragma unroll
      for (int ks = 0; ks < 12; ++ks)
        acc[tk] = wmma16(load_a_frag(arow + ks * 32),
                         load_b_frag(brow + ks * 32), acc[tk]);
    }
    __syncthreads();
  }

  // ---- out = y + mlp + b2, staged transposed for coalesced NCHW store ----
  for (int tk = 0; tk < 6; ++tk) {
    const int task = wid + tk * 8;
    const int mt = task / 12, nt = task % 12;
    const int ng = nt * 16 + nloc;
    const float bb = b2[ng];
#pragma unroll
    for (int i = 0; i < 8; ++i) {
      const int m = mt * 16 + m0 + i;
      outT[ng * 64 + m] = acc[tk][i] + bb + y[m * C_DIM + ng];
    }
  }
  __syncthreads();
  for (int idx = tid; idx < C_DIM * 64; idx += 256) {
    const int c = idx / 64, m = idx % 64;
    out[((size_t)b * C_DIM + c) * HWPLANE + hw0 + m] = outT[idx];
  }
}

// ---------------------------------------------------------------------------
extern "C" void kernel_launch(void* const* d_in, const int* in_sizes, int n_in,
                              void* d_out, int out_size, void* d_ws,
                              size_t ws_size, hipStream_t stream) {
  const float* x     = (const float*)d_in[0];
  const float* wq    = (const float*)d_in[1];
  const float* bq    = (const float*)d_in[2];
  const float* wk    = (const float*)d_in[3];
  const float* bk    = (const float*)d_in[4];
  const float* wv    = (const float*)d_in[5];
  const float* bv    = (const float*)d_in[6];
  const float* wo    = (const float*)d_in[7];
  const float* bo    = (const float*)d_in[8];
  const float* w1    = (const float*)d_in[9];
  const float* b1    = (const float*)d_in[10];
  const float* w2    = (const float*)d_in[11];
  const float* b2    = (const float*)d_in[12];
  const float* ln1_g = (const float*)d_in[13];
  const float* ln1_b = (const float*)d_in[14];
  const float* ln2_g = (const float*)d_in[15];
  const float* ln2_b = (const float*)d_in[16];
  // d_in[17] = shift_size (==3, folded into addressing)

  const size_t NTOK = (size_t)BATCH * HWPLANE;         // 200704
  char* ws = (char*)d_ws;
  _Float16* tn   = (_Float16*)ws;                       // 77,070,336 B
  float*    attn = (float*)(ws + NTOK * C_DIM * 2);     // 154,140,672 B
  char* wbase = ws + NTOK * C_DIM * 2 + NTOK * C_DIM * 4;
  _Float16* wqh = (_Float16*)wbase;
  _Float16* wkh = wqh + C_DIM * C_DIM;
  _Float16* wvh = wkh + C_DIM * C_DIM;
  _Float16* woh = wvh + C_DIM * C_DIM;
  _Float16* w1h = woh + C_DIM * C_DIM;
  _Float16* w2h = w1h + DF_DIM * C_DIM;

  const int nw = C_DIM * C_DIM, nf = DF_DIM * C_DIM;
  cast_f16_kernel<<<(nw + 255) / 256, 256, 0, stream>>>(wq, wqh, nw);
  cast_f16_kernel<<<(nw + 255) / 256, 256, 0, stream>>>(wk, wkh, nw);
  cast_f16_kernel<<<(nw + 255) / 256, 256, 0, stream>>>(wv, wvh, nw);
  cast_f16_kernel<<<(nw + 255) / 256, 256, 0, stream>>>(wo, woh, nw);
  cast_f16_kernel<<<(nf + 255) / 256, 256, 0, stream>>>(w1, w1h, nf);
  cast_f16_kernel<<<(nf + 255) / 256, 256, 0, stream>>>(w2, w2h, nf);

  ln1_kernel<<<BATCH * HW_IMG * (HW_IMG / 32), 256, 0, stream>>>(x, ln1_g,
                                                                 ln1_b, tn);

  const size_t attn_lds = (size_t)(5 * LPAD * C_DIM + 8 * 16 * LPAD) * 2;  // 139264
  swin_attn_kernel<<<BATCH * NWGRID * NWGRID, 256, attn_lds, stream>>>(
      tn, wqh, bq, wkh, bk, wvh, bv, woh, bo, attn);

  const size_t mlp_lds = (size_t)LPAD * C_DIM * 4 + (size_t)LPAD * C_DIM * 2 +
                         (size_t)LPAD * 384 * 2;                           // 122880
  mlp_kernel<<<(int)(NTOK / 64), 256, mlp_lds, stream>>>(
      x, attn, ln2_g, ln2_b, w1h, b1, w2h, b2, (float*)d_out);
}
